// VINN_22514218565723
// MI455X (gfx1250) — matrix operand order repaired and checked by reading, not compiled, not run on hardware
//
#include <hip/hip_runtime.h>
#include <stdint.h>

// ---------------------------------------------------------------------------
// VINN kNN-regression for MI455X (gfx1250, wave32, WMMA).
//   B=64, D=512, A=7, K=5 fixed by the reference; N derived from in_sizes.
// Split-precision bf16 GEMM (hi/lo, 3 WMMAs per K-slice) ~fp32 accuracy on the
// bf16 matrix pipe; single streaming pass over representations (HBM-bound,
// ~1.02 GB -> ~44 us at 23.3 TB/s). v_perm_b32 packing + v_fma_mix lo-split
// keeps the co-executing VALU stream below the WMMA+VMEM critical path.
// Workspace: 1 KB (qsq) + 128 KB (bf16 batch hi/lo) + ~5 MB (partial topk).
// ---------------------------------------------------------------------------

typedef __attribute__((ext_vector_type(16))) __bf16 v16bf;
typedef __attribute__((ext_vector_type(8)))  float  v8f;
typedef __attribute__((ext_vector_type(4)))  float  v4f;   // for NT loads

#define B_   64
#define D_   512
#define A_   7
#define K_   5
#define NT   256          // N columns per block
#define KS   32           // K per WMMA step
#define NSTEP (D_ / KS)   // 16

union AFrag { uint4 q[2]; v16bf v; };
union BPack { uint32_t w[8]; v16bf v; };

__device__ __forceinline__ void top5_insert(float v, int idx,
                                            float (&d)[K_], int (&ix)[K_]) {
  if (v < d[K_ - 1]) {
    d[K_ - 1] = v; ix[K_ - 1] = idx;
#pragma unroll
    for (int j = K_ - 1; j > 0; --j) {
      if (d[j] < d[j - 1]) {
        float td = d[j]; d[j] = d[j - 1]; d[j - 1] = td;
        int   ti = ix[j]; ix[j] = ix[j - 1]; ix[j - 1] = ti;
      }
    }
  }
}

// ------------------------- kernel 0: prep ----------------------------------
__global__ void vinn_prep(const float* __restrict__ brep,
                          float* __restrict__ qsq,
                          uint16_t* __restrict__ ahi,
                          uint16_t* __restrict__ alo) {
  __shared__ float red[128];
  const int r = blockIdx.x;
  const int t = threadIdx.x;
  float acc = 0.f;
  for (int d = t; d < D_; d += 128) {
    float x = brep[r * D_ + d];
    uint32_t u  = __float_as_uint(x);
    uint32_t hb = u & 0xFFFF0000u;
    float hf = __uint_as_float(hb);
    float lf = x - hf;                         // exact residual of truncation
    ahi[r * D_ + d] = (uint16_t)(hb >> 16);
    alo[r * D_ + d] = (uint16_t)(__float_as_uint(lf) >> 16);
    acc = fmaf(x, x, acc);
  }
  red[t] = acc;
  __syncthreads();
  for (int s = 64; s > 0; s >>= 1) {
    if (t < s) red[t] += red[t + s];
    __syncthreads();
  }
  if (t == 0) qsq[r] = red[0];
}

// ------------------ kernel 1: distances + per-block top-5 ------------------
// TAIL=false: all 256 columns in range (no guards, no exec juggling).
// TAIL=true : final partial block, column guards active.
template <bool TAIL>
__global__ void __launch_bounds__(256)
vinn_dist_topk(const float* __restrict__ reps, int n, int blkBase,
               const float* __restrict__ qsq,
               const uint16_t* __restrict__ ahi,
               const uint16_t* __restrict__ alo,
               float* __restrict__ part_d,
               int*   __restrict__ part_i) {
  __shared__ float sdist[B_ * NT];             // 64 KB: squared distances

  const int blk  = blkBase + blockIdx.x;
  const int tid  = threadIdx.x;
  const int lane = tid & 31;
  const int wid  = tid >> 5;
  const int rt   = wid & 3;                    // 4 row tiles of 16 (covers 64)
  const int ct   = wid >> 2;                   // 2 column halves of 128
  const int lh   = lane >> 4;                  // lane half (0/1)
  const int l15  = lane & 15;

  const int colBase = blk * NT + ct * 128;

  const v8f vzero = {0.f, 0.f, 0.f, 0.f, 0.f, 0.f, 0.f, 0.f};
  v8f   acc[8];
  float rsq[8];
#pragma unroll
  for (int c = 0; c < 8; ++c) { acc[c] = vzero; rsq[c] = 0.f; }

  // A fragment base pointers (ISA 16-bit A 16x32 layout: lanes 0-15 hold K
  // runs [k..k+7],[k+16..k+23]; lanes 16-31 shifted by 8)
  const int arow = rt * 16 + l15;
  const uint16_t* ah = ahi + arow * D_ + 8 * lh;
  const uint16_t* al = alo + arow * D_ + 8 * lh;

  for (int ks = 0; ks < NSTEP; ++ks) {
    const int kbase = ks * KS;

    AFrag a_hi, a_lo;
    a_hi.q[0] = *(const uint4*)(ah + kbase);
    a_hi.q[1] = *(const uint4*)(ah + kbase + 16);
    a_lo.q[0] = *(const uint4*)(al + kbase);
    a_lo.q[1] = *(const uint4*)(al + kbase + 16);

#pragma unroll
    for (int cc = 0; cc < 8; ++cc) {
      const int col = colBase + cc * 16 + l15;
      float f[16];
      if (!TAIL || col < n) {
        // B 32x16 layout: lanes 0-15 hold K=k..k+7,k+16..k+23; lanes 16-31 shifted by 8
        const float* bp = reps + (size_t)col * D_ + kbase + 8 * lh;
        v4f f0 = __builtin_nontemporal_load((const v4f*)bp);
        v4f f1 = __builtin_nontemporal_load((const v4f*)bp + 1);
        v4f f2 = __builtin_nontemporal_load((const v4f*)bp + 4);
        v4f f3 = __builtin_nontemporal_load((const v4f*)bp + 5);
        f[0]=f0.x; f[1]=f0.y; f[2]=f0.z;  f[3]=f0.w;
        f[4]=f1.x; f[5]=f1.y; f[6]=f1.z;  f[7]=f1.w;
        f[8]=f2.x; f[9]=f2.y; f[10]=f2.z; f[11]=f2.w;
        f[12]=f3.x;f[13]=f3.y;f[14]=f3.z; f[15]=f3.w;
        // Unconditional speculative prefetch of the next K slice (invalid
        // translations are silently dropped; avoids a per-cc scalar branch).
        if (!TAIL) __builtin_prefetch(bp + KS, 0, 3);
      } else {
#pragma unroll
        for (int e = 0; e < 16; ++e) f[e] = 0.f;
      }

      // Split each fp32 into bf16 hi (truncate) + bf16 lo (exact residual),
      // packing two elements per dword with v_perm_b32; lo lowers to
      // v_fma_mix_f32_bf16 (x - bf16(x)) on gfx1250.
      BPack b_hi, b_lo;
#pragma unroll
      for (int p = 0; p < 8; ++p) {
        const float fa = f[2 * p], fb = f[2 * p + 1];
        const uint32_t ua = __float_as_uint(fa);
        const uint32_t ub = __float_as_uint(fb);
        const float ha = __uint_as_float(ua & 0xFFFF0000u);
        const float hb = __uint_as_float(ub & 0xFFFF0000u);
        b_hi.w[p] = __builtin_amdgcn_perm(ub, ua, 0x07060302u);
        const float la = fa - ha;
        const float lb = fb - hb;
        b_lo.w[p] = __builtin_amdgcn_perm(__float_as_uint(lb),
                                          __float_as_uint(la), 0x07060302u);
        rsq[cc] = fmaf(fa, fa, fmaf(fb, fb, rsq[cc]));  // exact fp32 norms
      }

      // cross ~= hi*hi + hi*lo + lo*hi  (fp32 accumulate on matrix pipe)
      acc[cc] = __builtin_amdgcn_wmma_f32_16x16x32_bf16(
          false, a_hi.v, false, b_hi.v, (short)0, acc[cc], false, false);
      acc[cc] = __builtin_amdgcn_wmma_f32_16x16x32_bf16(
          false, a_hi.v, false, b_lo.v, (short)0, acc[cc], false, false);
      acc[cc] = __builtin_amdgcn_wmma_f32_16x16x32_bf16(
          false, a_lo.v, false, b_hi.v, (short)0, acc[cc], false, false);
    }
  }

  // q_sq values for the 8 rows this lane contributes to (C layout: M = j+8*lh)
  const int rbase = rt * 16 + 8 * lh;
  float qv[8];
#pragma unroll
  for (int j = 0; j < 8; ++j) qv[j] = qsq[rbase + j];

#pragma unroll
  for (int cc = 0; cc < 8; ++cc) {
    // column norm = this lane's K-half plus the opposite half (lane^16)
    float rs = rsq[cc] + __shfl_xor(rsq[cc], 16, 32);
    const int colLoc = ct * 128 + cc * 16 + l15;
#pragma unroll
    for (int j = 0; j < 8; ++j) {
      float d2 = fmaf(-2.f, acc[cc][j], qv[j] + rs);
      sdist[(rbase + j) * NT + colLoc] = fmaxf(d2, 0.f);
    }
  }
  __syncthreads();

  // ---- per-row top-5 over this block's 256 columns -----------------------
  // 4 threads per row, each scanning a rotated 64-column quarter.
  const int r = tid >> 2;                      // row 0..63
  const int q = tid & 3;                       // quarter 0..3
  float bd[K_] = {1e30f, 1e30f, 1e30f, 1e30f, 1e30f};
  int   bi[K_] = {0, 0, 0, 0, 0};
  const int gbase = blk * NT;
  for (int i = 0; i < 64; ++i) {
    const int c  = q * 64 + ((r + i) & 63);
    const int gc = gbase + c;
    float v = sdist[r * NT + c];
    if (TAIL && gc >= n) v = 1e30f;
    top5_insert(v, gc, bd, bi);
  }
  __syncthreads();                             // all sdist reads complete

  // Candidate buffer aliases the first 10 rows of sdist (already consumed).
  float* cd = sdist;                           // 256*5 floats
  int*   ci = (int*)(sdist + 256 * K_);        // 256*5 ints
#pragma unroll
  for (int j = 0; j < K_; ++j) { cd[tid * K_ + j] = bd[j]; ci[tid * K_ + j] = bi[j]; }
  __syncthreads();

  if (q == 0) {                                // one merger thread per row
    float fd[K_] = {1e30f, 1e30f, 1e30f, 1e30f, 1e30f};
    int   fi[K_] = {0, 0, 0, 0, 0};
#pragma unroll
    for (int s = 0; s < 4; ++s)
#pragma unroll
      for (int j = 0; j < K_; ++j) {
        const int cidx = (4 * r + s) * K_ + j;
        top5_insert(cd[cidx], ci[cidx], fd, fi);
      }
    float* pd = part_d + ((size_t)blk * B_ + r) * K_;
    int*   pi = part_i + ((size_t)blk * B_ + r) * K_;
#pragma unroll
    for (int j = 0; j < K_; ++j) { pd[j] = fd[j]; pi[j] = fi[j]; }
  }
}

// ---------------- kernel 2: global merge + softmax + gather ----------------
__global__ void vinn_merge(const float* __restrict__ part_d,
                           const int*   __restrict__ part_i,
                           const float* __restrict__ actions,
                           float* __restrict__ out, int nblk) {
  __shared__ float sd[128 * K_];
  __shared__ int   si[128 * K_];
  const int b = blockIdx.x;
  const int t = threadIdx.x;

  float bd[K_] = {1e30f, 1e30f, 1e30f, 1e30f, 1e30f};
  int   bi[K_] = {0, 0, 0, 0, 0};
  for (int blk = t; blk < nblk; blk += 128) {
    const float* pd = part_d + ((size_t)blk * B_ + b) * K_;
    const int*   pi = part_i + ((size_t)blk * B_ + b) * K_;
#pragma unroll
    for (int j = 0; j < K_; ++j) top5_insert(pd[j], pi[j], bd, bi);
  }
#pragma unroll
  for (int j = 0; j < K_; ++j) { sd[t * K_ + j] = bd[j]; si[t * K_ + j] = bi[j]; }
  __syncthreads();

  if (t == 0) {
    float fd[K_] = {1e30f, 1e30f, 1e30f, 1e30f, 1e30f};
    int   fi[K_] = {0, 0, 0, 0, 0};
    for (int c = 0; c < 128 * K_; ++c) top5_insert(sd[c], si[c], fd, fi);

    // softmax(-dist) over the 5 nearest; fd holds d^2, sorted ascending
    float d0 = sqrtf(fd[0]);
    float w[K_], s = 0.f;
#pragma unroll
    for (int j = 0; j < K_; ++j) { w[j] = __expf(-(sqrtf(fd[j]) - d0)); s += w[j]; }
    const float inv = 1.f / s;
#pragma unroll
    for (int a = 0; a < A_; ++a) {
      float o = 0.f;
#pragma unroll
      for (int j = 0; j < K_; ++j)
        o = fmaf(w[j], actions[(size_t)fi[j] * A_ + a], o);
      out[b * A_ + a] = o * inv;
    }
  }
}

// ------------------------------- launcher ----------------------------------
extern "C" void kernel_launch(void* const* d_in, const int* in_sizes, int n_in,
                              void* d_out, int out_size, void* d_ws, size_t ws_size,
                              hipStream_t stream) {
  const float* brep    = (const float*)d_in[0];   // [64, 512]
  const float* reps    = (const float*)d_in[1];   // [N, 512]
  const float* actions = (const float*)d_in[2];   // [N, 7]
  const int n     = in_sizes[1] / D_;
  const int nfull = n / NT;                       // guard-free blocks
  const int nblk  = (n + NT - 1) / NT;            // total blocks (incl. tail)

  char* ws = (char*)d_ws;
  float*    qsq    = (float*)ws;                                   // 64 f32
  uint16_t* ahi    = (uint16_t*)(ws + 1024);                       // 64x512 bf16
  uint16_t* alo    = (uint16_t*)(ws + 1024 + B_ * D_ * 2);         // 64x512 bf16
  float*    part_d = (float*)(ws + 1024 + 2 * B_ * D_ * 2);        // nblk*64*5
  int*      part_i = (int*)((char*)part_d + (size_t)nblk * B_ * K_ * sizeof(float));

  vinn_prep<<<B_, 128, 0, stream>>>(brep, qsq, ahi, alo);
  if (nfull > 0)
    vinn_dist_topk<false><<<nfull, 256, 0, stream>>>(reps, n, 0, qsq, ahi, alo,
                                                     part_d, part_i);
  if (nblk > nfull)
    vinn_dist_topk<true><<<1, 256, 0, stream>>>(reps, n, nfull, qsq, ahi, alo,
                                                part_d, part_i);
  vinn_merge<<<B_, 128, 0, stream>>>(part_d, part_i, actions, (float*)d_out, nblk);
}
